// LPIPS_SMIC_58626303590534
// MI455X (gfx1250) — matrix operand order repaired and checked by reading, not compiled, not run on hardware
//
#include <hip/hip_runtime.h>
#include <hip/hip_bf16.h>

typedef __attribute__((ext_vector_type(16))) _Float16 v16h;
typedef __attribute__((ext_vector_type(8)))  _Float16 v8h;
typedef __attribute__((ext_vector_type(8)))  float    v8f;

// ---------------------------------------------------------------------------
// f32 -> f16 convert
// ---------------------------------------------------------------------------
__global__ void cvt_f32_f16(const float* __restrict__ src, _Float16* __restrict__ dst, int n) {
    int i = blockIdx.x * 256 + threadIdx.x;
    if (i < n) dst[i] = (_Float16)src[i];
}

// ---------------------------------------------------------------------------
// 3x3 SAME conv + bias + ReLU as implicit GEMM on v_wmma_f32_16x16x32_f16.
//   A = weights [Co x (Ci*9)], K-contiguous rows -> two b128 loads per lane.
//   B = im2col panel double-buffered through LDS: per K-step the block
//       gathers WN panels of 32(K) x 16(N) f16 into buffer buf^1 while the
//       waves run WMMA out of buffer buf; one barrier per K-step.
//   Per-element gather state (ci, dy, dx, pixel coords, LDS slot) lives in
//   registers and advances incrementally per K-step (no division in the
//   steady state); addresses are 32-bit element indices clamped by cndmask
//   so every global access is an unconditional SGPR-base + voffset load.
//   8 waves = WM x WN tiles; 'active' is wave-uniform so EXEC is all-ones
//   at the WMMA. Inactive waves still participate in fill + barriers.
// ---------------------------------------------------------------------------
template <int WN>
__global__ void __launch_bounds__(256)
conv3x3_wmma(const _Float16* __restrict__ X, const _Float16* __restrict__ Wt,
             const float* __restrict__ bias, _Float16* __restrict__ Y,
             int Ci, int H, int Wd, int Co) {
    constexpr int WM        = 8 / WN;
    constexpr int FILLPER   = WN * 2;               // fill elements per thread
    constexpr int BUFSTRIDE = WN * 16 * 32;         // f16 elems per buffer
    __shared__ _Float16 Bt[2][WN][16][32];          // [buf][wn][n][k]

    const int HW     = H * Wd;
    const int Ktot   = Ci * 9;
    const int tilesM = Co >> 4;
    const int tilesN = (HW + 15) >> 4;
    const int wave   = threadIdx.y;
    const int wn     = wave % WN;
    const int wm     = wave / WN;
    const int tm     = blockIdx.y * WM + wm;
    const int tn     = blockIdx.x * WN + wn;
    const bool active = (tm < tilesM) && (tn < tilesN);   // wave-uniform
    const int lane = threadIdx.x;
    const int hl   = lane >> 4;
    const int l15  = lane & 15;
    const int tid  = wave * 32 + lane;

    const int am = tm * 16 + l15;
    const _Float16* wrow = Wt + (size_t)(active ? am : 0) * Ktot;
    const int  n      = tn * 16 + l15;
    const bool nvalid = active && (n < HW);
    const bool kAligned = ((Ktot & 31) == 0);

    // ---- per-thread fill-element state (divisions done once, here) ----
    int  f_pym1[FILLPER], f_pxm1[FILLPER], f_kl[FILLPER];
    int  f_ci[FILLPER], f_dy[FILLPER], f_dx[FILLPER];
    int  f_lds[FILLPER];
    bool f_nok[FILLPER];
    _Float16* ldsBase = &Bt[0][0][0][0];
#pragma unroll
    for (int u = 0; u < FILLPER; ++u) {
        int e    = tid + 256 * u;
        int wn_e = e >> 9;
        int rem  = e & 511;
        int nl   = rem >> 5;                        // pixel within tile
        int kl   = rem & 31;                        // k within step
        int ne   = (blockIdx.x * WN + wn_e) * 16 + nl;
        int pye  = ne / Wd;
        f_pym1[u] = pye - 1;
        f_pxm1[u] = (ne - pye * Wd) - 1;
        f_nok[u]  = (ne < HW);
        f_kl[u]   = kl;
        int ci0   = kl / 9;
        int r0    = kl - ci0 * 9;
        f_ci[u]   = ci0;
        f_dy[u]   = r0 / 3;
        f_dx[u]   = r0 - (r0 / 3) * 3;
        f_lds[u]  = (wn_e * 16 + nl) * 32 + kl;
    }

    // branch-free gather of one K-step panel into buffer `buf`
    auto fill = [&](int buf, int k0) {
#pragma unroll
        for (int u = 0; u < FILLPER; ++u) {
            int ci = f_ci[u], dy = f_dy[u], dx = f_dx[u];
            int yy = f_pym1[u] + dy;
            int xx = f_pxm1[u] + dx;
            bool ok = f_nok[u] & ((k0 + f_kl[u]) < Ktot) &
                      ((unsigned)yy < (unsigned)H) & ((unsigned)xx < (unsigned)Wd);
            int idx = ok ? (ci * HW + yy * Wd + xx) : 0;     // 32-bit index
            _Float16 v = X[idx];                             // unconditional load
            ldsBase[buf * BUFSTRIDE + f_lds[u]] = ok ? v : (_Float16)0.0f;
            // advance (ci, dy, dx) by k += 32  (32 = 3*9 + 5; 5 = 3*1 + 2)
            dx += 2; dy += 1; ci += 3;
            if (dx >= 3) { dx -= 3; dy += 1; }
            if (dy >= 3) { dy -= 3; ci += 1; }
            f_ci[u] = ci; f_dy[u] = dy; f_dx[u] = dx;
        }
    };

    v8f acc = {};
    fill(0, 0);
    __syncthreads();
    int buf = 0;
    for (int k0 = 0; k0 < Ktot; k0 += 32) {
        if (k0 + 32 < Ktot) {
            if (active) __builtin_prefetch((const void*)(wrow + k0 + 32), 0, 3);
            fill(buf ^ 1, k0 + 32);                 // overlap with WMMA below
        }
        if (active) {
            // A fragment: lanes 0-15 K=k0+{0..7,16..23}; 16-31 K=k0+{8..15,24..31}
            v16h a;
            if (kAligned) {
                v8h lo = *(const v8h*)(wrow + k0 + hl * 8);
                v8h hi = *(const v8h*)(wrow + k0 + hl * 8 + 16);
#pragma unroll
                for (int j = 0; j < 8; ++j) { a[j] = lo[j]; a[8 + j] = hi[j]; }
            } else {                                // layer 0: Ktot = 27
#pragma unroll
                for (int j = 0; j < 8; ++j) {
                    int k1 = k0 + hl * 8 + j;
                    int k2 = k1 + 16;
                    bool o1 = (k1 < Ktot), o2 = (k2 < Ktot);
                    _Float16 t1 = wrow[o1 ? k1 : 0];
                    _Float16 t2 = wrow[o2 ? k2 : 0];
                    a[j]     = o1 ? t1 : (_Float16)0.0f;
                    a[8 + j] = o2 ? t2 : (_Float16)0.0f;
                }
            }
            // B fragment from LDS (two contiguous b128 reads per lane)
            v8h b0 = *(const v8h*)&Bt[buf][wn][l15][hl * 8];
            v8h b1 = *(const v8h*)&Bt[buf][wn][l15][hl * 8 + 16];
            v16h b;
#pragma unroll
            for (int j = 0; j < 8; ++j) { b[j] = b0[j]; b[8 + j] = b1[j]; }

            acc = __builtin_amdgcn_wmma_f32_16x16x32_f16(
                false, a, false, b, (short)0, acc, false, false);
        }
        __syncthreads();
        buf ^= 1;
    }

    if (active) {
        // D VGPR r, lane l -> M = r + 8*hl, N = l15
#pragma unroll
        for (int r = 0; r < 8; ++r) {
            int m = tm * 16 + r + hl * 8;
            float v = acc[r] + bias[m];
            v = fmaxf(v, 0.0f);
            if (nvalid) Y[(size_t)m * HW + n] = (_Float16)v;
        }
    }
}

// ---------------------------------------------------------------------------
// 2x2 stride-2 max pool (f16)
// ---------------------------------------------------------------------------
__global__ void maxpool2x2(const _Float16* __restrict__ X, _Float16* __restrict__ Y,
                           int C, int H, int W) {
    int Ho = H >> 1, Wo = W >> 1;
    size_t tot = (size_t)C * Ho * Wo;
    size_t i = (size_t)blockIdx.x * 256 + threadIdx.x;
    if (i >= tot) return;
    int wo = (int)(i % Wo);
    size_t t = i / Wo;
    int ho = (int)(t % Ho);
    int c  = (int)(t / Ho);
    const _Float16* p = X + (size_t)c * H * W + (size_t)(2 * ho) * W + 2 * wo;
    float a = fmaxf(fmaxf((float)p[0], (float)p[1]), fmaxf((float)p[W], (float)p[W + 1]));
    Y[i] = (_Float16)a;
}

// ---------------------------------------------------------------------------
// LPIPS term: atomic-add  mean_{h,w} sum_c lw[c]*(A-B)^2  into out[0]
// ---------------------------------------------------------------------------
__global__ void lpips_term(const _Float16* __restrict__ A, const _Float16* __restrict__ B,
                           const float* __restrict__ lw, float* out, int C, int HW) {
    __shared__ float red[256];
    size_t tot = (size_t)C * HW;
    float s = 0.0f;
    for (size_t i = (size_t)blockIdx.x * 256 + threadIdx.x; i < tot;
         i += (size_t)gridDim.x * 256) {
        int c = (int)(i / (size_t)HW);
        float d = (float)A[i] - (float)B[i];
        s += lw[c] * d * d;
    }
    red[threadIdx.x] = s;
    __syncthreads();
    for (int o = 128; o > 0; o >>= 1) {
        if (threadIdx.x < (unsigned)o) red[threadIdx.x] += red[threadIdx.x + o];
        __syncthreads();
    }
    if (threadIdx.x == 0) atomicAdd(out, red[0] / (float)HW);
}

// ---------------------------------------------------------------------------
// 1x1 random projection to 32 channels (f32 out)
// ---------------------------------------------------------------------------
__global__ void project32(const _Float16* __restrict__ T, const float* __restrict__ PJ,
                          float* __restrict__ M, int C, int HW) {
    int idx = blockIdx.x * 256 + threadIdx.x;
    if (idx >= 32 * HW) return;
    int p  = idx % HW;
    int co = idx / HW;
    const float* w = PJ + (size_t)co * C;
    float s = 0.0f;
    for (int c = 0; c < C; ++c) s += w[c] * (float)T[(size_t)c * HW + p];
    M[idx] = s;
}

// ---------------------------------------------------------------------------
// MIC statistic on one (patch, channel) pair of 49 samples.
// Stable ranks, equi-frequency bins bx = rx*nx/49, MI with the reference's
// 1e-12 epsilon, best/ln(2). One 64-thread block per pair (blockIdx = p*32+ch).
// ---------------------------------------------------------------------------
__global__ void mic_kernel(const float* __restrict__ M0, const float* __restrict__ M1,
                           float* __restrict__ mic, int W, int PW) {
    int pair = blockIdx.x;
    int ch = pair & 31;
    int p  = pair >> 5;
    int HW = W * W;
    __shared__ float xs[49], ys[49];
    __shared__ int   rx[49], ry[49];
    int t = threadIdx.x;
    if (t < 49) {
        int pr = t / 7, pc = t - pr * 7;
        int ph = p / PW, pw = p - ph * PW;
        int pix = (ph * 7 + pr) * W + (pw * 7 + pc);
        xs[t] = M0[(size_t)ch * HW + pix];
        ys[t] = M1[(size_t)ch * HW + pix];
    }
    __syncthreads();
    if (t < 49) {
        int cx = 0, cy = 0;
        for (int j = 0; j < 49; ++j) {
            cx += (xs[j] < xs[t]) || (xs[j] == xs[t] && j < t);
            cy += (ys[j] < ys[t]) || (ys[j] == ys[t] && j < t);
        }
        rx[t] = cx;
        ry[t] = cy;
    }
    __syncthreads();
    if (t == 0) {
        const int GX[3] = {2, 2, 3}, GY[3] = {2, 3, 2};
        float best = 0.0f;
        for (int g = 0; g < 3; ++g) {
            int nx = GX[g], ny = GY[g];
            int cnt[9];
            for (int i = 0; i < 9; ++i) cnt[i] = 0;
            for (int i = 0; i < 49; ++i) {
                int bx = (rx[i] * nx) / 49;
                int by = (ry[i] * ny) / 49;
                cnt[bx * ny + by]++;
            }
            int pxc[3] = {0, 0, 0}, pyc[3] = {0, 0, 0};
            for (int bx = 0; bx < nx; ++bx)
                for (int by = 0; by < ny; ++by) {
                    pxc[bx] += cnt[bx * ny + by];
                    pyc[by] += cnt[bx * ny + by];
                }
            float mi = 0.0f;
            for (int bx = 0; bx < nx; ++bx)
                for (int by = 0; by < ny; ++by) {
                    float pxy = cnt[bx * ny + by] / 49.0f;
                    if (pxy > 0.0f) {
                        float px = pxc[bx] / 49.0f;
                        float py = pyc[by] / 49.0f;
                        mi += pxy * (logf(pxy + 1e-12f) - logf(px * py + 1e-12f));
                    }
                }
            best = fmaxf(best, mi / logf(2.0f));
        }
        mic[pair] = best;
    }
}

// ---------------------------------------------------------------------------
// Patch SSD: pd[p] = sum_{c,49} (T0 - T1)^2 over the 7x7 patch
// ---------------------------------------------------------------------------
__global__ void pdiff_kernel(const _Float16* __restrict__ T0, const _Float16* __restrict__ T1,
                             float* __restrict__ pd, int C, int W, int PW) {
    int p = blockIdx.x;
    int HW = W * W;
    int ph = p / PW, pw = p - ph * PW;
    __shared__ float red[256];
    float s = 0.0f;
    for (int i = threadIdx.x; i < C * 49; i += 256) {
        int j = i % 49, c = i / 49;
        int pr = j / 7, pc = j - pr * 7;
        size_t pix = (size_t)c * HW + (size_t)(ph * 7 + pr) * W + (pw * 7 + pc);
        float d = (float)T0[pix] - (float)T1[pix];
        s += d * d;
    }
    red[threadIdx.x] = s;
    __syncthreads();
    for (int o = 128; o > 0; o >>= 1) {
        if (threadIdx.x < (unsigned)o) red[threadIdx.x] += red[threadIdx.x + o];
        __syncthreads();
    }
    if (threadIdx.x == 0) pd[p] = red[0];
}

// ---------------------------------------------------------------------------
// SMIC combine: out[0] += mean_p (1 - mean_ch mic[p,ch]) * pd[p]
// ---------------------------------------------------------------------------
__global__ void smic_final(const float* __restrict__ mic, const float* __restrict__ pd,
                           float* out, int L) {
    __shared__ float red[64];
    float s = 0.0f;
    for (int p = threadIdx.x; p < L; p += 64) {
        float m = 0.0f;
        for (int ch = 0; ch < 32; ++ch) m += mic[p * 32 + ch];
        m *= (1.0f / 32.0f);
        s += (1.0f - m) * pd[p];
    }
    red[threadIdx.x] = s;
    __syncthreads();
    for (int o = 32; o > 0; o >>= 1) {
        if (threadIdx.x < (unsigned)o) red[threadIdx.x] += red[threadIdx.x + o];
        __syncthreads();
    }
    if (threadIdx.x == 0) atomicAdd(out, red[0] / (float)L);
}

__global__ void zero1(float* o) {
    if (blockIdx.x == 0 && threadIdx.x == 0) o[0] = 0.0f;
}

// ---------------------------------------------------------------------------
// Host orchestration
// ---------------------------------------------------------------------------
static const int CI_[13] = {3, 64, 64, 128, 128, 256, 256, 256, 512, 512, 512, 512, 512};
static const int CO_[13] = {64, 64, 128, 128, 256, 256, 256, 512, 512, 512, 512, 512, 512};
static const int HH_[13] = {224, 224, 112, 112, 56, 56, 56, 28, 28, 28, 14, 14, 14};

extern "C" void kernel_launch(void* const* d_in, const int* in_sizes, int n_in,
                              void* d_out, int out_size, void* d_ws, size_t ws_size,
                              hipStream_t stream) {
    (void)in_sizes; (void)n_in; (void)out_size; (void)ws_size;
    const float* in_img[2] = {(const float*)d_in[0], (const float*)d_in[1]};
    const float* vw[13];
    const float* vb[13];
    for (int i = 0; i < 13; ++i) { vw[i] = (const float*)d_in[2 + i]; vb[i] = (const float*)d_in[15 + i]; }
    const float* lw[3] = {(const float*)d_in[28], (const float*)d_in[29], (const float*)d_in[30]};
    const float* pj[2] = {(const float*)d_in[31], (const float*)d_in[32]};
    float* out = (float*)d_out;

    // bump allocator over d_ws
    size_t off = 0;
    auto alloc = [&](size_t bytes) -> void* {
        off = (off + 255) & ~(size_t)255;
        void* p = (char*)d_ws + off;
        off += bytes;
        return p;
    };

    // f16 weights
    _Float16* wf16[13];
    for (int i = 0; i < 13; ++i) {
        int n = CO_[i] * CI_[i] * 9;
        wf16[i] = (_Float16*)alloc((size_t)n * 2);
        cvt_f32_f16<<<(n + 255) / 256, 256, 0, stream>>>(vw[i], wf16[i], n);
    }

    // activations
    const size_t HW224 = 224 * 224;
    _Float16* Xin = (_Float16*)alloc(3 * HW224 * 2);
    _Float16* Ta  = (_Float16*)alloc(64 * HW224 * 2);
    _Float16* Tb  = (_Float16*)alloc(64 * HW224 * 2);
    _Float16* F[2][5];
    const size_t FElems[5] = {64 * HW224, (size_t)128 * 112 * 112, (size_t)256 * 56 * 56,
                              (size_t)512 * 28 * 28, (size_t)512 * 14 * 14};
    for (int im = 0; im < 2; ++im)
        for (int k = 0; k < 5; ++k) F[im][k] = (_Float16*)alloc(FElems[k] * 2);

    // SMIC scratch
    float* m2[2] = {(float*)alloc(32 * 3136 * 4), (float*)alloc(32 * 3136 * 4)};
    float* m3[2] = {(float*)alloc(32 * 784 * 4), (float*)alloc(32 * 784 * 4)};
    float* micbuf = (float*)alloc(64 * 32 * 4);
    float* pdbuf  = (float*)alloc(64 * 4);

    auto conv = [&](const _Float16* in, int li, _Float16* outp) {
        int H = HH_[li], Co = CO_[li];
        int HW = H * H;
        int tilesM = Co >> 4;
        int tilesN = (HW + 15) >> 4;
        dim3 blk(32, 8);
        if (tilesM >= 8) {
            dim3 grid(tilesN, (tilesM + 7) / 8);
            conv3x3_wmma<1><<<grid, blk, 0, stream>>>(in, wf16[li], vb[li], outp,
                                                      CI_[li], H, H, Co);
        } else {                                  // Co=64 layers: 4 M-waves x 2 N-tiles
            dim3 grid((tilesN + 1) / 2, (tilesM + 3) / 4);
            conv3x3_wmma<2><<<grid, blk, 0, stream>>>(in, wf16[li], vb[li], outp,
                                                      CI_[li], H, H, Co);
        }
    };
    auto pool = [&](const _Float16* in, _Float16* outp, int C, int H) {
        size_t tot = (size_t)C * (H / 2) * (H / 2);
        maxpool2x2<<<(unsigned)((tot + 255) / 256), 256, 0, stream>>>(in, outp, C, H, H);
    };

    for (int im = 0; im < 2; ++im) {
        cvt_f32_f16<<<(int)((3 * HW224 + 255) / 256), 256, 0, stream>>>(in_img[im], Xin,
                                                                        (int)(3 * HW224));
        conv(Xin, 0, Ta);
        conv(Ta, 1, F[im][0]);
        pool(F[im][0], Tb, 64, 224);
        conv(Tb, 2, Ta);
        conv(Ta, 3, F[im][1]);
        pool(F[im][1], Tb, 128, 112);
        conv(Tb, 4, Ta);
        conv(Ta, 5, Tb);
        conv(Tb, 6, F[im][2]);
        pool(F[im][2], Ta, 256, 56);
        conv(Ta, 7, Tb);
        conv(Tb, 8, Ta);
        conv(Ta, 9, F[im][3]);
        pool(F[im][3], Tb, 512, 28);
        conv(Tb, 10, Ta);
        conv(Ta, 11, Tb);
        conv(Tb, 12, F[im][4]);
    }

    // val = 0
    zero1<<<1, 1, 0, stream>>>(out);

    // LPIPS head: layers 0, 1, 4
    {
        const int kk[3] = {0, 1, 4};
        const int C[3] = {64, 128, 512};
        const int HW[3] = {224 * 224, 112 * 112, 14 * 14};
        for (int li = 0; li < 3; ++li) {
            size_t tot = (size_t)C[li] * HW[li];
            int grid = (int)((tot + 255) / 256);
            if (grid > 2048) grid = 2048;
            lpips_term<<<grid, 256, 0, stream>>>(F[0][kk[li]], F[1][kk[li]], lw[li], out,
                                                 C[li], HW[li]);
        }
    }

    // SMIC: layer 2 (256ch, 56x56, L=64) and layer 3 (512ch, 28x28, L=16)
    {
        project32<<<(32 * 3136 + 255) / 256, 256, 0, stream>>>(F[0][2], pj[0], m2[0], 256, 3136);
        project32<<<(32 * 3136 + 255) / 256, 256, 0, stream>>>(F[1][2], pj[0], m2[1], 256, 3136);
        mic_kernel<<<64 * 32, 64, 0, stream>>>(m2[0], m2[1], micbuf, 56, 8);
        pdiff_kernel<<<64, 256, 0, stream>>>(F[0][2], F[1][2], pdbuf, 256, 56, 8);
        smic_final<<<1, 64, 0, stream>>>(micbuf, pdbuf, out, 64);

        project32<<<(32 * 784 + 255) / 256, 256, 0, stream>>>(F[0][3], pj[1], m3[0], 512, 784);
        project32<<<(32 * 784 + 255) / 256, 256, 0, stream>>>(F[1][3], pj[1], m3[1], 512, 784);
        mic_kernel<<<16 * 32, 64, 0, stream>>>(m3[0], m3[1], micbuf, 28, 4);
        pdiff_kernel<<<16, 256, 0, stream>>>(F[0][3], F[1][3], pdbuf, 512, 28, 4);
        smic_final<<<1, 64, 0, stream>>>(micbuf, pdbuf, out, 16);
    }
}
